// MoEGPT_39745627357694
// MI455X (gfx1250) — matrix operand order repaired and checked by reading, not compiled, not run on hardware
//
#include <hip/hip_runtime.h>

// ---------------------------------------------------------------------------
// MoE-GPT forward for MI455X (gfx1250), wave32, WMMA bf16 16x16x32, f32 acc.
// Weight-BW-bound (~475MB fp32 weights @ 23.3TB/s ~ 20us): weights are read
// once as fp32 and packed to bf16 in-register via v_perm_b32 (1 op / 2 elems).
// Activation A-tiles are staged to LDS with the Tensor Data Mover
// (tensor_load_to_lds + s_wait_tensorcnt) to kill the 4x cross-wave re-read.
// Attention keeps a 64x1024 score tile in LDS, padded to a 1028-float row
// stride so softmax + fragment loads are bank-conflict-free (264KB < 320KB).
// MoE computed sparsely (top-2 only), deterministic routing (no atomics).
// ---------------------------------------------------------------------------

#define TT   1024
#define CC   768
#define C3   2304
#define HH   12
#define HDIM 64
#define EE   8
#define FF   3072
#define VV   50257
#define LL   2
#define SP   1028   // padded LDS row stride (floats) for the attention S tile

typedef __attribute__((ext_vector_type(16))) __bf16        v16bf;
typedef __attribute__((ext_vector_type(8)))  float         v8f;
typedef __attribute__((ext_vector_type(4)))  unsigned int  v4u;
typedef __attribute__((ext_vector_type(8)))  int           v8i;
typedef __attribute__((ext_vector_type(4)))  int           v4i;

struct alignas(16) U4 { unsigned int x, y, z, w; };
struct alignas(16) F4 { float x, y, z, w; };

union FragBF {
  v16bf v;
  unsigned short s[16];
  unsigned int u[8];
  U4 q[2];
};

#define PERM_PACK_HI16 0x07060302u  // dst = {src0[3:2], src1[3:2]} bytes

__device__ __forceinline__ unsigned short f32_to_bf16(float f) {
  unsigned int u = __float_as_uint(f);
  u += 0x7FFFu + ((u >> 16) & 1u);           // round-to-nearest-even
  return (unsigned short)(u >> 16);
}
__device__ __forceinline__ float bf16_to_f32(unsigned short h) {
  return __uint_as_float(((unsigned int)h) << 16);
}
__device__ __forceinline__ v8f vzero8() {
  v8f z = {0.f,0.f,0.f,0.f,0.f,0.f,0.f,0.f};
  return z;
}
__device__ __forceinline__ v8f wmma_bf16(v16bf a, v16bf b, v8f c) {
  return __builtin_amdgcn_wmma_f32_16x16x32_bf16(false, a, false, b, (short)0, c,
                                                 false, false);
}
// A fragment (16x32 bf16): lane holds row (lane&15); two 8-elem K chunks at
// kb and kb+16, kb = 8*(lane>>4).  p = rowbase + k0 + kb.
__device__ __forceinline__ v16bf load_afrag(const unsigned short* p) {
  FragBF f;
  f.q[0] = *(const U4*)(p);
  f.q[1] = *(const U4*)(p + 16);
  return f.v;
}
// B fragment (32x16 bf16): lane holds col (lane&15); two 8-elem K chunks at
// koff and koff+16, koff = 8*(lane>>4).  p = colbase(row of W) + k0 + koff.
__device__ __forceinline__ v16bf load_bfrag_bf16(const unsigned short* p) {
  FragBF f;
  f.q[0] = *(const U4*)(p);
  f.q[1] = *(const U4*)(p + 16);
  return f.v;
}
// f32 at p[0..7] and p[16..23] -> bf16 fragment.  Truncating pack: one v_perm_b32 per
// element pair (8 VALU total instead of ~48 for the RNE chain).
__device__ __forceinline__ v16bf load_bfrag_f32(const float* p) {
  const U4* q = (const U4*)p;
  U4 q0 = q[0], q1 = q[1], q2 = q[4], q3 = q[5];
  FragBF f;
  f.u[0] = __builtin_amdgcn_perm(q0.y, q0.x, PERM_PACK_HI16);
  f.u[1] = __builtin_amdgcn_perm(q0.w, q0.z, PERM_PACK_HI16);
  f.u[2] = __builtin_amdgcn_perm(q1.y, q1.x, PERM_PACK_HI16);
  f.u[3] = __builtin_amdgcn_perm(q1.w, q1.z, PERM_PACK_HI16);
  f.u[4] = __builtin_amdgcn_perm(q2.y, q2.x, PERM_PACK_HI16);
  f.u[5] = __builtin_amdgcn_perm(q2.w, q2.z, PERM_PACK_HI16);
  f.u[6] = __builtin_amdgcn_perm(q3.y, q3.x, PERM_PACK_HI16);
  f.u[7] = __builtin_amdgcn_perm(q3.w, q3.z, PERM_PACK_HI16);
  return f.v;
}
__device__ __forceinline__ float gelu_exact(float x) {
  return 0.5f * x * (1.0f + erff(x * 0.70710678118654752f));
}

// Stage 64 rows x Kd cols of bf16 A into LDS.  TDM path: one DMA descriptor
// issued by wave 0 (D# per CDNA5 ISA), tracked by
// TENSORcnt; fallback is a cooperative b128 copy loop.
#if __has_builtin(__builtin_amdgcn_tensor_load_to_lds)
#pragma message("CDNA5 probe: tensor_load_to_lds builtin PRESENT - TDM path enabled")
#else
#pragma message("CDNA5 probe: tensor_load_to_lds builtin ABSENT - cooperative-copy fallback")
#endif
__device__ __forceinline__ void stage_A_tile(const unsigned short* Aglob,
                                             unsigned short* Alds, int Kd) {
#if __has_builtin(__builtin_amdgcn_tensor_load_to_lds)
  if (threadIdx.x < 32) {
    unsigned long long ga = (unsigned long long)(size_t)(const void*)Aglob;
    unsigned int lds = (unsigned int)(size_t)(void*)Alds;  // LDS byte offset
    v4u g0;
    g0.x = 1u;                                   // count=1 valid descriptor
    g0.y = lds;                                  // lds_addr
    g0.z = (unsigned int)ga;                     // global_addr[31:0]
    g0.w = (unsigned int)(ga >> 32) | (2u << 30);// global_addr[56:32] | type=2
    v8i g1;
    g1[0] = (int)(1u << 16);                               // data_size=2B
    g1[1] = (int)(((unsigned)Kd & 0xFFFFu) << 16);         // tensor_dim0 lo16
    g1[2] = (int)((((unsigned)Kd >> 16) & 0xFFFFu) | ((unsigned)TT << 16)); // dim0 hi | dim1 lo
    g1[3] = (int)(((unsigned)Kd & 0xFFFFu) << 16);         // dim1 hi(0) | tile_dim0=Kd
    g1[4] = 64;                                            // tile_dim1=64 rows
    g1[5] = Kd;                                            // tensor_dim0_stride lo32
    g1[6] = 0;
    g1[7] = 0;
    v4i z4 = {0, 0, 0, 0};
#if defined(__clang_major__) && (__clang_major__ >= 23)
    v8i z8 = {0, 0, 0, 0, 0, 0, 0, 0};
    __builtin_amdgcn_tensor_load_to_lds(g0, g1, z4, z4, z8, 0);
#else
    __builtin_amdgcn_tensor_load_to_lds(g0, g1, z4, z4, 0);
#endif
    __builtin_amdgcn_s_wait_tensorcnt(0);
  }
  __syncthreads();
#else
  const U4* src = (const U4*)Aglob;
  U4* dst = (U4*)Alds;
  int nchunk = 64 * Kd / 8;
  for (int i = threadIdx.x; i < nchunk; i += 256) dst[i] = src[i];
  __syncthreads();
#endif
}

// ---------------------------------------------------------------------------
// Embedding: x[t][c] = wte[idx[t]][c] + wpe[t][c]
// ---------------------------------------------------------------------------
__global__ void k_embed(const int* __restrict__ idx, const float* __restrict__ wte,
                        const float* __restrict__ wpe, float* __restrict__ x) {
  int t = blockIdx.x;
  int row = idx[t];
  for (int c = threadIdx.x; c < CC; c += blockDim.x)
    x[(size_t)t * CC + c] = wte[(size_t)row * CC + c] + wpe[(size_t)t * CC + c];
}

// ---------------------------------------------------------------------------
// LayerNorm (one row per block). Writes bf16 (GEMM input) and optional fp32.
// ---------------------------------------------------------------------------
__global__ void k_layernorm(const float* __restrict__ X, const float* __restrict__ g,
                            unsigned short* __restrict__ obf, float* __restrict__ of32,
                            int rowoff) {
  __shared__ float red[256];
  int tid = threadIdx.x;
  int r = blockIdx.x + rowoff;
  const float* xr = X + (size_t)r * CC;
  float s = 0.f;
  for (int c = tid; c < CC; c += 256) s += xr[c];
  red[tid] = s; __syncthreads();
  for (int off = 128; off > 0; off >>= 1) {
    if (tid < off) red[tid] += red[tid + off];
    __syncthreads();
  }
  float mu = red[0] * (1.0f / CC);
  __syncthreads();
  s = 0.f;
  for (int c = tid; c < CC; c += 256) { float d = xr[c] - mu; s += d * d; }
  red[tid] = s; __syncthreads();
  for (int off = 128; off > 0; off >>= 1) {
    if (tid < off) red[tid] += red[tid + off];
    __syncthreads();
  }
  float rstd = rsqrtf(red[0] * (1.0f / CC) + 1e-5f);
  for (int c = tid; c < CC; c += 256) {
    float v = (xr[c] - mu) * rstd * g[c];
    obf[(size_t)blockIdx.x * CC + c] = f32_to_bf16(v);
    if (of32) of32[(size_t)blockIdx.x * CC + c] = v;
  }
}

// ---------------------------------------------------------------------------
// Generic GEMM: O[M,N](f32) (+)= A[M,K](bf16) * W[N,K](f32)^T
// Block: 256 thr = 8 waves as 2x4; block tile 64x128, wave tile 32x32.
// A block-tile staged to LDS via TDM; B streamed from global with prefetch.
// Dynamic LDS: 64*Kd*2 bytes.
// ---------------------------------------------------------------------------
__global__ __launch_bounds__(256) void k_gemm(const unsigned short* __restrict__ A,
                                              const float* __restrict__ W,
                                              float* __restrict__ O,
                                              int N, int Kd, int accumulate) {
  extern __shared__ unsigned short Atile[];   // [64][Kd] bf16
  int m0b = blockIdx.y * 64;
  stage_A_tile(A + (size_t)m0b * Kd, Atile, Kd);

  int w = threadIdx.x >> 5, lane = threadIdx.x & 31;
  int wm = (w >> 2) * 32;                      // wave row offset inside tile
  int n0 = blockIdx.x * 128 + (w & 3) * 32;
  int kb = (lane >> 4) << 3;
  int koff = (lane >> 4) << 3;
  const unsigned short* pa0 = Atile + (size_t)(wm + (lane & 15)) * Kd + kb;
  const unsigned short* pa1 = Atile + (size_t)(wm + 16 + (lane & 15)) * Kd + kb;
  const float* pw0 = W + (size_t)(n0 + (lane & 15)) * Kd + koff;
  const float* pw1 = W + (size_t)(n0 + 16 + (lane & 15)) * Kd + koff;
  v8f c00 = vzero8(), c01 = vzero8(), c10 = vzero8(), c11 = vzero8();
  for (int k0 = 0; k0 < Kd; k0 += 32) {
    __builtin_prefetch(pw0 + k0 + 64, 0, 1);
    __builtin_prefetch(pw1 + k0 + 64, 0, 1);
    v16bf a0 = load_afrag(pa0 + k0);
    v16bf a1 = load_afrag(pa1 + k0);
    v16bf b0 = load_bfrag_f32(pw0 + k0);
    v16bf b1 = load_bfrag_f32(pw1 + k0);
    c00 = wmma_bf16(a0, b0, c00);
    c01 = wmma_bf16(a0, b1, c01);
    c10 = wmma_bf16(a1, b0, c10);
    c11 = wmma_bf16(a1, b1, c11);
  }
  int m0 = m0b + wm;
  int nc = lane & 15, mh = (lane >> 4) << 3;
#pragma unroll
  for (int e = 0; e < 8; ++e) {
    int mA = m0 + e + mh;
    int mB = mA + 16;
    float* oa = O + (size_t)mA * N + n0 + nc;
    float* ob = O + (size_t)mB * N + n0 + nc;
    if (accumulate) {
      oa[0] += c00[e]; oa[16] += c01[e];
      ob[0] += c10[e]; ob[16] += c11[e];
    } else {
      oa[0] = c00[e]; oa[16] = c01[e];
      ob[0] = c10[e]; ob[16] = c11[e];
    }
  }
}

// ---------------------------------------------------------------------------
// QKV split: fp32 [T,3C] -> bf16 q[H][T][64] (pre-scaled 1/8), k[H][T][64],
// vT[H][64][T] (transposed so P@V B-fragments are K-contiguous).
// ---------------------------------------------------------------------------
__global__ void k_split_qkv(const float* __restrict__ qkv, unsigned short* __restrict__ qh,
                            unsigned short* __restrict__ kh, unsigned short* __restrict__ vt) {
  int t = blockIdx.x;
  for (int i = threadIdx.x; i < CC; i += blockDim.x) {
    int h = i >> 6, d = i & 63;
    const float* base = qkv + (size_t)t * C3;
    float qv = base[i] * 0.125f;           // fold 1/sqrt(64)
    float kv = base[CC + i];
    float vv = base[2 * CC + i];
    qh[((size_t)h * TT + t) * HDIM + d] = f32_to_bf16(qv);
    kh[((size_t)h * TT + t) * HDIM + d] = f32_to_bf16(kv);
    vt[((size_t)h * HDIM + d) * TT + t] = f32_to_bf16(vv);
  }
}

// ---------------------------------------------------------------------------
// Attention: one block per (head, 64-query tile).
// LDS: S[64][SP] f32 (SP=1028 pad -> conflict-free banks) + red[256].
// Phase 1: S = (Q/8)K^T via WMMA, causal-masked. Phase 2: softmax in LDS.
// Phase 3: O = P @ V via WMMA (A from LDS, B from transposed V in global).
// ---------------------------------------------------------------------------
__global__ __launch_bounds__(256) void k_attn(const unsigned short* __restrict__ Q,
                                              const unsigned short* __restrict__ Kb,
                                              const unsigned short* __restrict__ Vt,
                                              unsigned short* __restrict__ Y) {
  extern __shared__ float smem[];
  float* S = smem;                    // [64][SP]
  float* red = smem + 64 * SP;        // [256]
  int h = blockIdx.x;
  int q0 = blockIdx.y * 64;
  int tid = threadIdx.x;
  int w = tid >> 5, lane = tid & 31;
  const unsigned short* Qh = Q + (size_t)h * TT * HDIM;
  const unsigned short* Kh = Kb + (size_t)h * TT * HDIM;
  const unsigned short* Vh = Vt + (size_t)h * HDIM * TT;
  int kb = (lane >> 4) << 3;
  int koff = (lane >> 4) << 3;

  { // ---- phase 1: scores ----
    int qsub = w >> 1, khalf = w & 1;
    int qr = q0 + qsub * 16 + (lane & 15);
    const unsigned short* pq = Qh + (size_t)qr * HDIM + kb;
    v16bf a0 = load_afrag(pq);
    v16bf a1 = load_afrag(pq + 32);
    int qlast = q0 + 63;
    for (int kt = khalf * 512; kt < khalf * 512 + 512; kt += 16) {
      int n = kt + (lane & 15);
      if (kt > qlast) {
#pragma unroll
        for (int e2 = 0; e2 < 8; ++e2) {
          int m = qsub * 16 + e2 + ((lane >> 4) << 3);
          S[m * SP + n] = -3.0e38f;
        }
      } else {
        const unsigned short* pk = Kh + (size_t)(kt + (lane & 15)) * HDIM + koff;
        v16bf b0 = load_bfrag_bf16(pk);
        v16bf b1 = load_bfrag_bf16(pk + 32);
        v8f acc = vzero8();
        acc = wmma_bf16(a0, b0, acc);
        acc = wmma_bf16(a1, b1, acc);
#pragma unroll
        for (int e2 = 0; e2 < 8; ++e2) {
          int m = qsub * 16 + e2 + ((lane >> 4) << 3);
          S[m * SP + n] = (n <= q0 + m) ? acc[e2] : -3.0e38f;
        }
      }
    }
  }
  __syncthreads();
  { // ---- phase 2: softmax (4 threads per row) ----
    int row = tid >> 2, sub = tid & 3;
    float* Sr = S + row * SP;
    float mx = -3.0e38f;
    for (int i = sub; i < 1024; i += 4) mx = fmaxf(mx, Sr[i]);
    red[tid] = mx; __syncthreads();
    mx = fmaxf(fmaxf(red[row * 4 + 0], red[row * 4 + 1]),
               fmaxf(red[row * 4 + 2], red[row * 4 + 3]));
    __syncthreads();
    float sm = 0.f;
    for (int i = sub; i < 1024; i += 4) {
      float p = __expf(Sr[i] - mx);
      Sr[i] = p; sm += p;
    }
    red[tid] = sm; __syncthreads();
    sm = red[row * 4 + 0] + red[row * 4 + 1] + red[row * 4 + 2] + red[row * 4 + 3];
    float inv = 1.0f / sm;
    for (int i = sub; i < 1024; i += 4) Sr[i] *= inv;
  }
  __syncthreads();
  { // ---- phase 3: O = P @ V ----
    int qsub = w & 3, hh = w >> 2;
    int mrow = qsub * 16;
    v8f o0 = vzero8(), o1 = vzero8();
    int kkend = q0 + mrow + 16;
    kkend = (kkend + 31) & ~31;
    if (kkend > 1024) kkend = 1024;
    for (int kk = 0; kk < kkend; kk += 32) {
      // A-frag from LDS probabilities, perm-packed f32 -> bf16
      const float* sp = S + (mrow + (lane & 15)) * SP + kk + kb;
      U4 s0 = *(const U4*)(sp);
      U4 s1 = *(const U4*)(sp + 4);
      U4 s2 = *(const U4*)(sp + 16);
      U4 s3 = *(const U4*)(sp + 20);
      FragBF fa;
      fa.u[0] = __builtin_amdgcn_perm(s0.y, s0.x, PERM_PACK_HI16);
      fa.u[1] = __builtin_amdgcn_perm(s0.w, s0.z, PERM_PACK_HI16);
      fa.u[2] = __builtin_amdgcn_perm(s1.y, s1.x, PERM_PACK_HI16);
      fa.u[3] = __builtin_amdgcn_perm(s1.w, s1.z, PERM_PACK_HI16);
      fa.u[4] = __builtin_amdgcn_perm(s2.y, s2.x, PERM_PACK_HI16);
      fa.u[5] = __builtin_amdgcn_perm(s2.w, s2.z, PERM_PACK_HI16);
      fa.u[6] = __builtin_amdgcn_perm(s3.y, s3.x, PERM_PACK_HI16);
      fa.u[7] = __builtin_amdgcn_perm(s3.w, s3.z, PERM_PACK_HI16);
      const unsigned short* pv0 = Vh + (size_t)(hh * 32 + (lane & 15)) * TT + kk + koff;
      const unsigned short* pv1 = Vh + (size_t)(hh * 32 + 16 + (lane & 15)) * TT + kk + koff;
      v16bf b0 = load_bfrag_bf16(pv0);
      v16bf b1 = load_bfrag_bf16(pv1);
      o0 = wmma_bf16(fa.v, b0, o0);
      o1 = wmma_bf16(fa.v, b1, o1);
    }
#pragma unroll
    for (int e2 = 0; e2 < 8; ++e2) {
      int m = mrow + e2 + ((lane >> 4) << 3);
      int t = q0 + m;
      int d0 = hh * 32 + (lane & 15);
      Y[(size_t)t * CC + h * HDIM + d0] = f32_to_bf16(o0[e2]);
      Y[(size_t)t * CC + h * HDIM + d0 + 16] = f32_to_bf16(o1[e2]);
    }
  }
}

// ---------------------------------------------------------------------------
// Router: logits[t][e] = dot(xn[t], gate[e])  (thread per (t,e) pair)
// ---------------------------------------------------------------------------
__global__ void k_router(const unsigned short* __restrict__ Xn, const float* __restrict__ G,
                         float* __restrict__ logits) {
  int tid = blockIdx.x * blockDim.x + threadIdx.x;
  if (tid >= TT * EE) return;
  int t = tid >> 3, e = tid & 7;
  const unsigned short* xr = Xn + (size_t)t * CC;
  const float* g = G + (size_t)e * CC;
  float s = 0.f;
  for (int c = 0; c < CC; ++c) s += bf16_to_f32(xr[c]) * g[c];
  logits[tid] = s;
}

// Top-2 + softmax over the two. F[e][t] = weight if selected else -1.
__global__ void k_topk(const float* __restrict__ logits, int* __restrict__ sel,
                       float* __restrict__ selw, float* __restrict__ F) {
  int t = blockIdx.x * blockDim.x + threadIdx.x;
  if (t >= TT) return;
  const float* lg = logits + (size_t)t * EE;
  int i0 = 0; float v0 = lg[0];
  for (int e = 1; e < EE; ++e) if (lg[e] > v0) { v0 = lg[e]; i0 = e; }
  int i1 = -1; float v1 = -3.0e38f;
  for (int e = 0; e < EE; ++e) if (e != i0 && lg[e] > v1) { v1 = lg[e]; i1 = e; }
  float e1 = __expf(v1 - v0);
  float inv = 1.0f / (1.0f + e1);
  float w0 = inv, w1 = e1 * inv;
  sel[t * 2] = i0; sel[t * 2 + 1] = i1;
  selw[t * 2] = w0; selw[t * 2 + 1] = w1;
  for (int e = 0; e < EE; ++e)
    F[(size_t)e * TT + t] = (e == i0) ? w0 : ((e == i1) ? w1 : -1.0f);
}

// Deterministic per-expert compaction (serial per expert; trivial work).
__global__ void k_scan(const float* __restrict__ F, int* __restrict__ perm,
                       int* __restrict__ tokslot, int* __restrict__ cnt) {
  int e = blockIdx.x;
  if (threadIdx.x != 0) return;
  int c = 0;
  for (int t = 0; t < TT; ++t) {
    float w = F[(size_t)e * TT + t];
    if (w >= 0.0f) {
      perm[(size_t)e * TT + c] = t;
      tokslot[(size_t)e * TT + t] = c;
      ++c;
    } else {
      tokslot[(size_t)e * TT + t] = 0;
    }
  }
  cnt[e] = c;
}

// ---------------------------------------------------------------------------
// MoE FFN1: H[e][slot][:] = gelu( xn[perm[e][slot]] @ w1_e^T ), bf16 out.
// ---------------------------------------------------------------------------
__global__ __launch_bounds__(256) void k_moe_ffn1(const unsigned short* __restrict__ Xn,
                                                  const float* __restrict__ W1,
                                                  const int* __restrict__ perm,
                                                  const int* __restrict__ cnt,
                                                  unsigned short* __restrict__ Hout) {
  int e = blockIdx.z;
  int ce = cnt[e];
  if ((int)(blockIdx.y * 64) >= ce) return;
  int w = threadIdx.x >> 5, lane = threadIdx.x & 31;
  int m0 = blockIdx.y * 64 + (w >> 2) * 32;
  int n0 = blockIdx.x * 128 + (w & 3) * 32;
  int kb = (lane >> 4) << 3, koff = (lane >> 4) << 3;
  int s0 = m0 + (lane & 15);       if (s0 > ce - 1) s0 = ce - 1;
  int s1 = m0 + 16 + (lane & 15);  if (s1 > ce - 1) s1 = ce - 1;
  int t0 = perm[(size_t)e * TT + s0];
  int t1 = perm[(size_t)e * TT + s1];
  const unsigned short* pa0 = Xn + (size_t)t0 * CC + kb;
  const unsigned short* pa1 = Xn + (size_t)t1 * CC + kb;
  const float* Wb = W1 + (size_t)e * FF * CC;
  const float* pw0 = Wb + (size_t)(n0 + (lane & 15)) * CC + koff;
  const float* pw1 = Wb + (size_t)(n0 + 16 + (lane & 15)) * CC + koff;
  v8f c00 = vzero8(), c01 = vzero8(), c10 = vzero8(), c11 = vzero8();
  for (int k0 = 0; k0 < CC; k0 += 32) {
    __builtin_prefetch(pw0 + k0 + 64, 0, 1);
    __builtin_prefetch(pw1 + k0 + 64, 0, 1);
    v16bf a0 = load_afrag(pa0 + k0);
    v16bf a1 = load_afrag(pa1 + k0);
    v16bf b0 = load_bfrag_f32(pw0 + k0);
    v16bf b1 = load_bfrag_f32(pw1 + k0);
    c00 = wmma_bf16(a0, b0, c00);
    c01 = wmma_bf16(a0, b1, c01);
    c10 = wmma_bf16(a1, b0, c10);
    c11 = wmma_bf16(a1, b1, c11);
  }
  int nc = lane & 15, mh = (lane >> 4) << 3;
#pragma unroll
  for (int e2 = 0; e2 < 8; ++e2) {
    int mA = m0 + e2 + mh;
    int mB = mA + 16;
    if (mA < ce) {
      unsigned short* o = Hout + ((size_t)e * TT + mA) * FF + n0 + nc;
      o[0]  = f32_to_bf16(gelu_exact(c00[e2]));
      o[16] = f32_to_bf16(gelu_exact(c01[e2]));
    }
    if (mB < ce) {
      unsigned short* o = Hout + ((size_t)e * TT + mB) * FF + n0 + nc;
      o[0]  = f32_to_bf16(gelu_exact(c10[e2]));
      o[16] = f32_to_bf16(gelu_exact(c11[e2]));
    }
  }
}

// ---------------------------------------------------------------------------
// MoE FFN2: Y[e][slot][:] = H[e][slot] @ w2_e^T   (fp32 out, no weighting)
// ---------------------------------------------------------------------------
__global__ __launch_bounds__(256) void k_moe_ffn2(const unsigned short* __restrict__ Hin,
                                                  const float* __restrict__ W2,
                                                  const int* __restrict__ cnt,
                                                  float* __restrict__ Y) {
  int e = blockIdx.z;
  int ce = cnt[e];
  if ((int)(blockIdx.y * 64) >= ce) return;
  int w = threadIdx.x >> 5, lane = threadIdx.x & 31;
  int m0 = blockIdx.y * 64 + (w >> 2) * 32;
  int n0 = blockIdx.x * 128 + (w & 3) * 32;
  int kb = (lane >> 4) << 3, koff = (lane >> 4) << 3;
  int s0 = m0 + (lane & 15);       if (s0 > ce - 1) s0 = ce - 1;
  int s1 = m0 + 16 + (lane & 15);  if (s1 > ce - 1) s1 = ce - 1;
  const unsigned short* pa0 = Hin + ((size_t)e * TT + s0) * FF + kb;
  const unsigned short* pa1 = Hin + ((size_t)e * TT + s1) * FF + kb;
  const float* Wb = W2 + (size_t)e * CC * FF;
  const float* pw0 = Wb + (size_t)(n0 + (lane & 15)) * FF + koff;
  const float* pw1 = Wb + (size_t)(n0 + 16 + (lane & 15)) * FF + koff;
  v8f c00 = vzero8(), c01 = vzero8(), c10 = vzero8(), c11 = vzero8();
  for (int k0 = 0; k0 < FF; k0 += 32) {
    __builtin_prefetch(pw0 + k0 + 64, 0, 1);
    __builtin_prefetch(pw1 + k0 + 64, 0, 1);
    v16bf a0 = load_afrag(pa0 + k0);
    v16bf a1 = load_afrag(pa1 + k0);
    v16bf b0 = load_bfrag_f32(pw0 + k0);
    v16bf b1 = load_bfrag_f32(pw1 + k0);
    c00 = wmma_bf16(a0, b0, c00);
    c01 = wmma_bf16(a0, b1, c01);
    c10 = wmma_bf16(a1, b0, c10);
    c11 = wmma_bf16(a1, b1, c11);
  }
  int nc = lane & 15, mh = (lane >> 4) << 3;
#pragma unroll
  for (int e2 = 0; e2 < 8; ++e2) {
    int mA = m0 + e2 + mh;
    int mB = mA + 16;
    if (mA < ce) {
      float* o = Y + ((size_t)e * TT + mA) * CC + n0 + nc;
      o[0] = c00[e2]; o[16] = c01[e2];
    }
    if (mB < ce) {
      float* o = Y + ((size_t)e * TT + mB) * CC + n0 + nc;
      o[0] = c10[e2]; o[16] = c11[e2];
    }
  }
}

// Combine: x[t] += w0*Y[e0][slot0] + w1*Y[e1][slot1]  (deterministic)
__global__ void k_moe_combine(float* __restrict__ X, const float* __restrict__ Y,
                              const int* __restrict__ sel, const float* __restrict__ selw,
                              const int* __restrict__ tokslot) {
  int t = blockIdx.x;
  int e0 = sel[t * 2], e1 = sel[t * 2 + 1];
  float w0 = selw[t * 2], w1 = selw[t * 2 + 1];
  int sl0 = tokslot[(size_t)e0 * TT + t];
  int sl1 = tokslot[(size_t)e1 * TT + t];
  const float* y0 = Y + ((size_t)e0 * TT + sl0) * CC;
  const float* y1 = Y + ((size_t)e1 * TT + sl1) * CC;
  for (int c = threadIdx.x; c < CC; c += blockDim.x)
    X[(size_t)t * CC + c] += w0 * y0[c] + w1 * y1[c];
}

// ---------------------------------------------------------------------------
// LM head GEMV (memory-bound: 154MB of wte): wave per vocab row, b128 loads.
// ---------------------------------------------------------------------------
__global__ void k_lmhead(const float* __restrict__ xnf, const float* __restrict__ wte,
                         float* __restrict__ out) {
  int w = threadIdx.x >> 5, lane = threadIdx.x & 31;
  int row = blockIdx.x * 8 + w;
  if (row >= VV) return;
  const float* wr = wte + (size_t)row * CC;
  float s = 0.f;
  for (int c = lane * 4; c < CC; c += 128) {
    F4 a = *(const F4*)(xnf + c);
    F4 b = *(const F4*)(wr + c);
    s += a.x * b.x + a.y * b.y + a.z * b.z + a.w * b.w;
  }
#pragma unroll
  for (int m = 16; m > 0; m >>= 1) s += __shfl_xor(s, m, 32);
  if (lane == 0) out[row] = s;
}

// ---------------------------------------------------------------------------
extern "C" void kernel_launch(void* const* d_in, const int* in_sizes, int n_in,
                              void* d_out, int out_size, void* d_ws, size_t ws_size,
                              hipStream_t stream) {
  (void)in_sizes; (void)n_in; (void)out_size; (void)ws_size;
  const int*   idx    = (const int*)  d_in[0];
  const float* wte    = (const float*)d_in[1];
  const float* wpe    = (const float*)d_in[2];
  const float* ln1_g  = (const float*)d_in[3];
  const float* qkv_w  = (const float*)d_in[4];
  const float* proj_w = (const float*)d_in[5];
  const float* ln2_g  = (const float*)d_in[6];
  const float* gate_w = (const float*)d_in[7];
  const float* w1     = (const float*)d_in[8];
  const float* w2     = (const float*)d_in[9];
  const float* lnf_g  = (const float*)d_in[10];
  float* out = (float*)d_out;

  char* ws = (char*)d_ws;
  size_t off = 0;
  auto alloc = [&](size_t bytes) -> void* {
    void* p = ws + off;
    off = (off + bytes + 255) & ~(size_t)255;
    return p;
  };
  float*          x       = (float*)         alloc((size_t)TT * CC * 4);
  unsigned short* xn      = (unsigned short*)alloc((size_t)TT * CC * 2);
  float*          qkv     = (float*)         alloc((size_t)TT * C3 * 4);
  unsigned short* qh      = (unsigned short*)alloc((size_t)HH * TT * HDIM * 2);
  unsigned short* kh      = (unsigned short*)alloc((size_t)HH * TT * HDIM * 2);
  unsigned short* vt      = (unsigned short*)alloc((size_t)HH * HDIM * TT * 2);
  unsigned short* ybf     = (unsigned short*)alloc((size_t)TT * CC * 2);
  float*          logits  = (float*)         alloc((size_t)TT * EE * 4);
  int*            sel     = (int*)           alloc((size_t)TT * 2 * 4);
  float*          selw    = (float*)         alloc((size_t)TT * 2 * 4);
  float*          Fmat    = (float*)         alloc((size_t)EE * TT * 4);
  int*            perm    = (int*)           alloc((size_t)EE * TT * 4);
  int*            tokslot = (int*)           alloc((size_t)EE * TT * 4);
  int*            cnt     = (int*)           alloc((size_t)EE * 4);
  unsigned short* hbuf    = (unsigned short*)alloc((size_t)EE * TT * FF * 2);
  float*          ymoe    = (float*)         alloc((size_t)EE * TT * CC * 4);
  unsigned short* xnf_bf  = (unsigned short*)alloc((size_t)CC * 2);
  float*          xnf     = (float*)         alloc((size_t)CC * 4);

  size_t smem_attn = (size_t)(64 * SP + 256) * sizeof(float);  // ~264KB < 320KB/WGP
  size_t smem_gemm = (size_t)64 * CC * 2;                      // 96KB A tile

  k_embed<<<TT, 256, 0, stream>>>(idx, wte, wpe, x);

  for (int l = 0; l < LL; ++l) {
    // ---- attention block ----
    k_layernorm<<<TT, 256, 0, stream>>>(x, ln1_g + (size_t)l * CC, xn, nullptr, 0);
    k_gemm<<<dim3(C3 / 128, TT / 64), 256, smem_gemm, stream>>>(
        xn, qkv_w + (size_t)l * C3 * CC, qkv, C3, CC, 0);
    k_split_qkv<<<TT, 256, 0, stream>>>(qkv, qh, kh, vt);
    k_attn<<<dim3(HH, TT / 64), 256, smem_attn, stream>>>(qh, kh, vt, ybf);
    k_gemm<<<dim3(CC / 128, TT / 64), 256, smem_gemm, stream>>>(
        ybf, proj_w + (size_t)l * CC * CC, x, CC, CC, 1 /*residual add*/);
    // ---- MoE block ----
    k_layernorm<<<TT, 256, 0, stream>>>(x, ln2_g + (size_t)l * CC, xn, nullptr, 0);
    k_router<<<(TT * EE + 255) / 256, 256, 0, stream>>>(
        xn, gate_w + (size_t)l * EE * CC, logits);
    k_topk<<<(TT + 255) / 256, 256, 0, stream>>>(logits, sel, selw, Fmat);
    k_scan<<<EE, 32, 0, stream>>>(Fmat, perm, tokslot, cnt);
    k_moe_ffn1<<<dim3(FF / 128, TT / 64, EE), 256, 0, stream>>>(
        xn, w1 + (size_t)l * EE * FF * CC, perm, cnt, hbuf);
    k_moe_ffn2<<<dim3(CC / 128, TT / 64, EE), 256, 0, stream>>>(
        hbuf, w2 + (size_t)l * EE * CC * FF, cnt, ymoe);
    k_moe_combine<<<TT, 256, 0, stream>>>(x, ymoe, sel, selw, tokslot);
  }

  // final LN on last position only + tied lm head
  k_layernorm<<<1, 256, 0, stream>>>(x, lnf_g, xnf_bf, xnf, TT - 1);
  k_lmhead<<<(VV + 7) / 8, 256, 0, stream>>>(xnf, wte, out);
}
